// RVSA_MRAM_61340722922141
// MI455X (gfx1250) — matrix operand order repaired and checked by reading, not compiled, not run on hardware
//
#include <hip/hip_runtime.h>
#include <hip/hip_bf16.h>

typedef __attribute__((ext_vector_type(16))) _Float16 v16h;
typedef __attribute__((ext_vector_type(4)))  _Float16 v4h;
typedef __attribute__((ext_vector_type(8)))  float    v8f;

// ---------------- problem constants ----------------
constexpr int cB = 2, cC = 512, cH = 128, cW = 128, cK = 32;
constexpr int cNH = 8, cNPT = 16; // cNPT = B*NH
// windows: 4x4 of 32x32, S=1024, nW=16

// Packed-B layout: for matrix (Kd x N), element (k,n) lives at
//   ((kc*(N/16) + jg)*32 + g*16 + ln)*16 + hi
// with kc=k>>5, g=(k>>4)&1, hi=k&15, jg=n>>4, ln=n&15.
__device__ __host__ inline long long packB_idx(int k, int n, int N) {
  int kc = k >> 5, g = (k >> 4) & 1, hi = k & 15;
  int jg = n >> 4, ln = n & 15;
  return (((long long)kc * (N >> 4) + jg) * 32 + g * 16 + ln) * 16 + hi;
}

// ---------------- WMMA fragment loads from LDS (contiguous k = 16*(lane/16) + s in A and B) ----
// A tile in LDS: row-major [128][36] halves. Per lane: two contiguous 16B runs.
__device__ inline v16h frag_a_lds(const _Float16* base, int ld) {
  int lane = threadIdx.x & 31;
  int g = lane >> 4, m = lane & 15;
  const _Float16* row = base + m * ld + 16 * g;
  const v4h* p0 = (const v4h*)(row);       // k = 16g .. 16g+7
  const v4h* p1 = (const v4h*)(row + 8);   // k = 16g+8 .. 16g+15
  v4h a = p0[0], b = p0[1], c = p1[0], d = p1[1];
  v16h f;
#pragma unroll
  for (int q = 0; q < 4; ++q) { f[q] = a[q]; f[4 + q] = b[q]; f[8 + q] = c[q]; f[12 + q] = d[q]; }
  return f;
}
// B tile in LDS: fragment-packed, 20-half (40B) lane stride, conflict-free.
__device__ inline v16h frag_b_packed(const _Float16* Bp, int j) {
  int lane = threadIdx.x & 31;
  const v4h* p = (const v4h*)(Bp + (j * 32 + lane) * 20);
  v4h a = p[0], b = p[1], c = p[2], d = p[3];
  v16h f;
#pragma unroll
  for (int q = 0; q < 4; ++q) { f[q] = a[q]; f[4 + q] = b[q]; f[8 + q] = c[q]; f[12 + q] = d[q]; }
  return f;
}

// ---------------- generic batched WMMA GEMM: C = alpha*A*B (+bias) --------
// A: row-major f16 (M x Kd), B: fragment-packed f16 (Kd x N), C: f32/f16.
#define GF_F16OUT 1
#define GF_BIAS   2
#define GF_TRANS  4

__global__ __launch_bounds__(256) void gemm_wmma_kernel(
    const _Float16* __restrict__ Ag, const _Float16* __restrict__ Bg,
    void* __restrict__ Cp, const float* __restrict__ bias,
    int M, int N, int Kd, int lda, long long ldc,
    int batInner, long long aIn, long long aOut,
    long long bIn, long long bOut, long long cIn, long long cOut,
    int flags, float alpha)
{
  __shared__ _Float16 As[128][36];
  __shared__ _Float16 Bp[8 * 32 * 20];
  int nb = blockIdx.z;
  const _Float16* Ab = Ag + (long long)(nb % batInner) * aIn + (long long)(nb / batInner) * aOut;
  const _Float16* Bb = Bg + (long long)(nb % batInner) * bIn + (long long)(nb / batInner) * bOut;
  long long cBase = (long long)(nb % batInner) * cIn + (long long)(nb / batInner) * cOut;

  int m0 = blockIdx.x * 128, n0 = blockIdx.y * 128;
  int tid = threadIdx.x, wave = tid >> 5, lane = tid & 31;
  int wm = wave >> 1, wn = wave & 1;
  int nGroups = N >> 4, jgBase = n0 >> 4;

  v8f acc[2][4];
#pragma unroll
  for (int i = 0; i < 2; ++i)
#pragma unroll
    for (int j = 0; j < 4; ++j) acc[i][j] = v8f{0.f,0.f,0.f,0.f,0.f,0.f,0.f,0.f};

  for (int k0 = 0; k0 < Kd; k0 += 32) {
    { // A tile 128x32, 2 threads/row x 16 halves
      int r = tid >> 1, c0 = (tid & 1) * 16;
      int gm = m0 + r;
      v4h z = v4h{(_Float16)0.f,(_Float16)0.f,(_Float16)0.f,(_Float16)0.f};
      v4h va = z, vb = z, vc = z, vd = z;
      if (gm < M) {
        const v4h* s = (const v4h*)(Ab + (long long)gm * lda + (k0 + c0));
        va = s[0]; vb = s[1]; vc = s[2]; vd = s[3];
      }
      v4h* d = (v4h*)(&As[r][c0]);
      d[0] = va; d[1] = vb; d[2] = vc; d[3] = vd;
    }
    { // B tile: straight copy of packed blocks (32B per thread)
      int slot = tid >> 5;            // 0..7 = nGroup within block
      int ln32 = tid & 31;            // packed lane
      int srcJg = jgBase + slot;
      int kcI = k0 >> 5;
      v4h z = v4h{(_Float16)0.f,(_Float16)0.f,(_Float16)0.f,(_Float16)0.f};
      v4h va = z, vb = z, vc = z, vd = z;
      if (srcJg < nGroups) {
        const v4h* s = (const v4h*)(Bb + (((long long)kcI * nGroups + srcJg) * 32 + ln32) * 16);
        va = s[0]; vb = s[1]; vc = s[2]; vd = s[3];
      }
      v4h* d = (v4h*)(&Bp[(slot * 32 + ln32) * 20]);
      d[0] = va; d[1] = vb; d[2] = vc; d[3] = vd;
    }
    __syncthreads();
    v16h af[2], bf[4];
#pragma unroll
    for (int i = 0; i < 2; ++i) af[i] = frag_a_lds(&As[wm * 32 + i * 16][0], 36);
#pragma unroll
    for (int j = 0; j < 4; ++j) bf[j] = frag_b_packed(Bp, wn * 4 + j);
#pragma unroll
    for (int i = 0; i < 2; ++i)
#pragma unroll
      for (int j = 0; j < 4; ++j)
        acc[i][j] = __builtin_amdgcn_wmma_f32_16x16x32_f16(
            false, af[i], false, bf[j], (short)0, acc[i][j], false, false);
    __syncthreads();
  }

  int g = lane >> 4, ln = lane & 15;
#pragma unroll
  for (int i = 0; i < 2; ++i)
#pragma unroll
    for (int j = 0; j < 4; ++j)
#pragma unroll
      for (int r = 0; r < 8; ++r) {
        int m = m0 + wm * 32 + i * 16 + r + 8 * g;
        int n = n0 + wn * 64 + j * 16 + ln;
        if (m < M && n < N) {
          float v = acc[i][j][r] * alpha;
          if (flags & GF_BIAS) v += bias[n];
          long long o = (flags & GF_TRANS) ? (cBase + (long long)n * ldc + m)
                                           : (cBase + (long long)m * ldc + n);
          if (flags & GF_F16OUT) ((_Float16*)Cp)[o] = (_Float16)v;
          else                   ((float*)Cp)[o]    = v;
        }
      }
}

// ---------------- implicit-GEMM 3x3 conv + BN affine + ReLU ---------------
// Tile: one image row (W=128 pixels) x 128 out channels. K = 9*IC.
// Weights fragment-packed (9*IC x OC).
__global__ __launch_bounds__(256) void conv3x3_wmma_kernel(
    const _Float16* __restrict__ in0, const _Float16* __restrict__ in1, // NHWC, chan-stride 512 each
    const _Float16* __restrict__ Wt,
    const float* __restrict__ bnS, const float* __restrict__ bnT,
    float* __restrict__ outNCHW, _Float16* __restrict__ outNHWC,
    int IC, int Bn, int H, int W, int OC)
{
  __shared__ _Float16 As[128][36];
  __shared__ _Float16 Bp[8 * 32 * 20];
  int row = blockIdx.x;
  int b = row / H, y = row % H;
  int oc0 = blockIdx.y * 128;
  int tid = threadIdx.x, wave = tid >> 5, lane = tid & 31;
  int wm = wave >> 1, wn = wave & 1;
  int nGroups = OC >> 4, jgBase = oc0 >> 4;

  v8f acc[2][4];
#pragma unroll
  for (int i = 0; i < 2; ++i)
#pragma unroll
    for (int j = 0; j < 4; ++j) acc[i][j] = v8f{0.f,0.f,0.f,0.f,0.f,0.f,0.f,0.f};

  int chunksPerTap = IC >> 5;
  int kChunks = 9 * chunksPerTap;
  for (int kc = 0; kc < kChunks; ++kc) {
    int tap = kc / chunksPerTap;
    int i0 = (kc - tap * chunksPerTap) * 32;
    int dy = tap / 3 - 1, dx = tap % 3 - 1;
    int yy = y + dy;
    const _Float16* src; int ci;
    if (in1 && i0 >= 512) { src = in1; ci = i0 - 512; } else { src = in0; ci = i0; }
    { // A tile: 128 pixels of this row, 32 channels of tap
      int px = tid >> 1, c0 = (tid & 1) * 16;
      int xx = px + dx;
      bool valid = (yy >= 0) && (yy < H) && (xx >= 0) && (xx < W);
      v4h z = v4h{(_Float16)0.f,(_Float16)0.f,(_Float16)0.f,(_Float16)0.f};
      v4h va = z, vb = z, vc = z, vd = z;
      if (valid) {
        const v4h* s = (const v4h*)(src + ((long long)(b * H + yy) * W + xx) * 512 + ci + c0);
        va = s[0]; vb = s[1]; vc = s[2]; vd = s[3];
      }
      v4h* d = (v4h*)(&As[px][c0]);
      d[0] = va; d[1] = vb; d[2] = vc; d[3] = vd;
    }
    { // B tile: packed weight copy
      int slot = tid >> 5, ln32 = tid & 31;
      int srcJg = jgBase + slot;
      const v4h* s = (const v4h*)(Wt + (((long long)kc * nGroups + srcJg) * 32 + ln32) * 16);
      v4h va = s[0], vb = s[1], vc = s[2], vd = s[3];
      v4h* d = (v4h*)(&Bp[(slot * 32 + ln32) * 20]);
      d[0] = va; d[1] = vb; d[2] = vc; d[3] = vd;
    }
    __syncthreads();
    v16h af[2], bf[4];
#pragma unroll
    for (int i = 0; i < 2; ++i) af[i] = frag_a_lds(&As[wm * 32 + i * 16][0], 36);
#pragma unroll
    for (int j = 0; j < 4; ++j) bf[j] = frag_b_packed(Bp, wn * 4 + j);
#pragma unroll
    for (int i = 0; i < 2; ++i)
#pragma unroll
      for (int j = 0; j < 4; ++j)
        acc[i][j] = __builtin_amdgcn_wmma_f32_16x16x32_f16(
            false, af[i], false, bf[j], (short)0, acc[i][j], false, false);
    __syncthreads();
  }

  int g = lane >> 4, ln = lane & 15;
#pragma unroll
  for (int i = 0; i < 2; ++i)
#pragma unroll
    for (int j = 0; j < 4; ++j)
#pragma unroll
      for (int r = 0; r < 8; ++r) {
        int px = wm * 32 + i * 16 + r + 8 * g;
        int oc = oc0 + wn * 64 + j * 16 + ln;
        float v = acc[i][j][r] * bnS[oc] + bnT[oc];
        v = fmaxf(v, 0.f);
        if (outNHWC) outNHWC[((long long)(b * H + y) * W + px) * OC + oc] = (_Float16)v;
        if (outNCHW) outNCHW[(((long long)b * OC + oc) * H + y) * W + px] = v;
      }
}

// ---------------- helper / glue kernels -----------------------------------
__global__ void nchw_to_nhwc_f16(const float* __restrict__ in, _Float16* __restrict__ out,
                                 int Bn, int C, int H, int W) {
  long long i = (long long)blockIdx.x * blockDim.x + threadIdx.x;
  long long total = (long long)Bn * C * H * W;
  if (i >= total) return;
  int c = (int)(i % C); long long p = i / C;
  int x = (int)(p % W); long long p2 = p / W;
  int y = (int)(p2 % H); int b = (int)(p2 / H);
  out[i] = (_Float16)in[(((long long)b * C + c) * H + y) * W + x];
}

// weight (O,I) f32 -> fragment-packed f16 (Kd=I, N=O)
__global__ void wTpack_f16(const float* __restrict__ in, _Float16* __restrict__ out, int O, int I) {
  int t = blockIdx.x * blockDim.x + threadIdx.x;
  if (t >= O * I) return;
  int n = t % O, k = t / O;
  out[packB_idx(k, n, O)] = (_Float16)in[(long long)n * I + k];
}

// conv weight (O,I,3,3) f32 -> fragment-packed f16 (Kd=9I with k=tap*I+i, N=O)
__global__ void convwTpack_f16(const float* __restrict__ in, _Float16* __restrict__ out, int O, int I) {
  long long t = (long long)blockIdx.x * blockDim.x + threadIdx.x;
  long long total = (long long)9 * I * O;
  if (t >= total) return;
  int o = (int)(t % O); long long r = t / O;
  int i = (int)(r % I); int tap = (int)(r / I);
  int k = tap * I + i;
  out[packB_idx(k, o, O)] = (_Float16)in[(((long long)o * I + i) * 9) + tap];
}

__global__ void bn_params_kernel(const float* g, const float* bb, const float* m,
                                 const float* v, float* s, float* t, int C) {
  int i = blockIdx.x * blockDim.x + threadIdx.x;
  if (i >= C) return;
  float sc = g[i] * rsqrtf(v[i] + 1e-5f);
  s[i] = sc; t[i] = bb[i] - m[i] * sc;
}

// softmax over H*W per (b,k); logits/probs layout ((b*HW + p), k), in-place
__global__ __launch_bounds__(256) void probs_softmax_kernel(float* logits, int K, int HW) {
  __shared__ float red[256];
  int b = blockIdx.x / K, k = blockIdx.x % K;
  float* base = logits + (long long)b * HW * K + k;
  float mx = -1e30f;
  for (int p = threadIdx.x; p < HW; p += 256) mx = fmaxf(mx, base[(long long)p * K]);
  red[threadIdx.x] = mx; __syncthreads();
  for (int s = 128; s > 0; s >>= 1) {
    if (threadIdx.x < s) red[threadIdx.x] = fmaxf(red[threadIdx.x], red[threadIdx.x + s]);
    __syncthreads();
  }
  mx = red[0]; __syncthreads();
  float sum = 0.f;
  for (int p = threadIdx.x; p < HW; p += 256) sum += expf(base[(long long)p * K] - mx);
  red[threadIdx.x] = sum; __syncthreads();
  for (int s = 128; s > 0; s >>= 1) {
    if (threadIdx.x < s) red[threadIdx.x] += red[threadIdx.x + s];
    __syncthreads();
  }
  float inv = 1.f / red[0]; __syncthreads();
  for (int p = threadIdx.x; p < HW; p += 256)
    base[(long long)p * K] = expf(base[(long long)p * K] - mx) * inv;
}

__global__ void pooled_kernel(const float* __restrict__ x, float* __restrict__ pooled) {
  int idx = blockIdx.x * 256 + threadIdx.x;           // (b,c,i,j)
  if (idx >= cB * cC * 16) return;
  int j = idx & 3, i = (idx >> 2) & 3;
  int c = (idx >> 4) % cC, b = idx / (cC * 16);
  const float* base = x + (((long long)b * cC + c) * cH + i * 32) * cW + j * 32;
  float s = 0.f;
  for (int dy = 0; dy < 32; ++dy)
    for (int dx = 0; dx < 32; ++dx) s += base[dy * cW + dx];
  float m = s * (1.f / 1024.f);
  pooled[idx] = m > 0.f ? m : 0.01f * m;
}

// per (b,h,wy,wx): 5 params {off0,off1,scl0,scl1,ang}
__global__ void heads_kernel(const float* __restrict__ pooled,
                             const float* ow, const float* ob,
                             const float* sw, const float* sb,
                             const float* aw, const float* ab, float* hp) {
  int idx = blockIdx.x * 64 + threadIdx.x;            // (b,h,i,j)
  if (idx >= cB * cNH * 16) return;
  int j = idx & 3, i = (idx >> 2) & 3;
  int h = (idx >> 4) & 7, b = idx >> 7;
  const float* pb = pooled + (long long)b * cC * 16 + i * 4 + j;
  float a0 = ob[h * 2], a1 = ob[h * 2 + 1], a2 = sb[h * 2], a3 = sb[h * 2 + 1], a4 = ab[h];
  for (int c = 0; c < cC; ++c) {
    float pv = pb[c * 16];
    a0 += pv * ow[(h * 2) * cC + c];
    a1 += pv * ow[(h * 2 + 1) * cC + c];
    a2 += pv * sw[(h * 2) * cC + c];
    a3 += pv * sw[(h * 2 + 1) * cC + c];
    a4 += pv * aw[h * cC + c];
  }
  float* o = hp + (long long)idx * 5;
  o[0] = a0 * (1.f / 32.f); o[1] = a1 * (1.f / 32.f); o[2] = a2; o[3] = a3; o[4] = a4;
}

__global__ void grid_kernel(const float* __restrict__ hp, float* __restrict__ grid) {
  long long idx = (long long)blockIdx.x * 256 + threadIdx.x;  // (n,y,x)
  if (idx >= (long long)cNPT * cH * cW) return;
  int x = (int)(idx % cW), y = (int)((idx / cW) % cH), n = (int)(idx / (cW * cH));
  int wy = y >> 5, iy = y & 31, wx = x >> 5, ix = x & 31;
  const float* p = hp + ((long long)n * 16 + wy * 4 + wx) * 5;
  float off0 = p[0], off1 = p[1], s0 = p[2], s1 = p[3], a = p[4];
  float bwv = (ix - 15.5f) * (2.f / 127.f);
  float bhv = (iy - 15.5f) * (2.f / 127.f);
  float wc0 = bwv * (s0 + 1.f), wc1 = bhv * (s1 + 1.f);
  float sa = sinf(a), ca = cosf(a);
  float r0 = -wc1 * sa + wc0 * ca;
  float r1 =  wc1 * ca + wc0 * sa;
  float wrw = -1.f + 2.f * (wx * 32 + 15.5f) / 127.f;
  float wrh = -1.f + 2.f * (wy * 32 + 15.5f) / 127.f;
  grid[idx * 2]     = wrw + r0 + off0;
  grid[idx * 2 + 1] = wrh + r1 + off1;
}

// bilinear-sample key features -> tx, written fragment-packed per batch f:
// batch f=(n*16+w), Kd = s (1024), N = d (64)
__global__ __launch_bounds__(256) void sample_tx_kernel(const float* __restrict__ keyf,
                                                        const float* __restrict__ grid,
                                                        _Float16* __restrict__ txo) {
  int d = threadIdx.x & 63, sq = threadIdx.x >> 6;
  int s = blockIdx.x * 4 + sq, w = blockIdx.y, n = blockIdx.z;
  int b = n >> 3, h = n & 7;
  int wy = w >> 2, wx = w & 3;
  int y = wy * 32 + (s >> 5), x = wx * 32 + (s & 31);
  const float* g = grid + (((long long)n * cH + y) * cW + x) * 2;
  float gx = (g[0] + 1.f) * 0.5f * (cW - 1);
  float gy = (g[1] + 1.f) * 0.5f * (cH - 1);
  float x0 = floorf(gx), y0 = floorf(gy);
  float fx = gx - x0, fy = gy - y0;
  int ix0 = (int)x0, iy0 = (int)y0;
  int chan = h * 64 + d;
  float acc = 0.f;
#pragma unroll
  for (int ty = 0; ty < 2; ++ty)
#pragma unroll
    for (int txp = 0; txp < 2; ++txp) {
      int xi = ix0 + txp, yi = iy0 + ty;
      float wgt = (txp ? fx : 1.f - fx) * (ty ? fy : 1.f - fy);
      if (xi >= 0 && xi < cW && yi >= 0 && yi < cH)
        acc += keyf[(((long long)b * cH + yi) * cW + xi) * cC + chan] * wgt;
    }
  long long f = (long long)n * 16 + w;
  txo[f * 65536 + packB_idx(s, d, 64)] = (_Float16)acc;
}

// bilinear-sample probs (replicates reference tile quirk probs[n % B]);
// writes TRANSPOSED tpT[(f*32 + k), s] (A-side of lc GEMM, row-major)
__global__ __launch_bounds__(256) void sample_tp_kernel(const float* __restrict__ probs,
                                                        const float* __restrict__ grid,
                                                        _Float16* __restrict__ tpo) {
  int k = threadIdx.x & 31, sq = threadIdx.x >> 5;
  int s = blockIdx.x * 8 + sq, w = blockIdx.y, n = blockIdx.z;
  int wy = w >> 2, wx = w & 3;
  int y = wy * 32 + (s >> 5), x = wx * 32 + (s & 31);
  int bp = n % cB;
  const float* g = grid + (((long long)n * cH + y) * cW + x) * 2;
  float gx = (g[0] + 1.f) * 0.5f * (cW - 1);
  float gy = (g[1] + 1.f) * 0.5f * (cH - 1);
  float x0 = floorf(gx), y0 = floorf(gy);
  float fx = gx - x0, fy = gy - y0;
  int ix0 = (int)x0, iy0 = (int)y0;
  float acc = 0.f;
#pragma unroll
  for (int ty = 0; ty < 2; ++ty)
#pragma unroll
    for (int txp = 0; txp < 2; ++txp) {
      int xi = ix0 + txp, yi = iy0 + ty;
      float wgt = (txp ? fx : 1.f - fx) * (ty ? fy : 1.f - fy);
      if (xi >= 0 && xi < cW && yi >= 0 && yi < cH)
        acc += probs[(((long long)bp * cH + yi) * cW + xi) * cK + k] * wgt;
    }
  long long f = (long long)n * 16 + w;
  tpo[(f * 32 + k) * 1024 + s] = (_Float16)acc;
}

// lc scramble (reference reshape quirk): lc2 batch (bn,h2) <- lcT[f=b*128+w'*8+h2]
// lc2 is B-side of dots: per batch Kd=d (64), N=k (32); fragment-packed.
__global__ void lc2_kernel(const _Float16* __restrict__ lcT, _Float16* __restrict__ lc2) {
  int idx = blockIdx.x * 256 + threadIdx.x;
  if (idx >= 256 * 2048) return;
  int k = idx & 31, d = (idx >> 5) & 63;
  int h2 = (idx >> 11) & 7, bn = idx >> 14;
  int b = bn >> 4, wp = bn & 15;
  int f = b * 128 + wp * 8 + h2;
  lc2[(long long)(bn * 8 + h2) * 2048 + packB_idx(d, k, 32)] =
      lcT[((long long)f * 64 + d) * 32 + k];
}

// window+head reorder of q: qw[(bn,h,s,d)] from q NHWC (A-side, row-major)
__global__ void qw_kernel(const _Float16* __restrict__ qpix, _Float16* __restrict__ qw) {
  long long idx = (long long)blockIdx.x * 256 + threadIdx.x;
  if (idx >= (long long)32 * 8 * 1024 * 64) return;
  int d = (int)(idx & 63);
  int s = (int)((idx >> 6) & 1023);
  int h = (int)((idx >> 16) & 7);
  int bn = (int)(idx >> 19);
  int b = bn >> 4, wy = (bn >> 2) & 3, wx = bn & 3;
  int y = wy * 32 + (s >> 5), x = wx * 32 + (s & 31);
  qw[idx] = qpix[(((long long)b * cH + y) * cW + x) * cC + h * 64 + d];
}

// per-row softmax over K=32 (one wave per row)
__global__ __launch_bounds__(256) void attn_softmax_kernel(const float* __restrict__ dots,
                                                           _Float16* __restrict__ attn) {
  long long row = (long long)blockIdx.x * 8 + (threadIdx.x >> 5);
  int lane = threadIdx.x & 31;
  float v = dots[row * 32 + lane];
  float mx = v;
  for (int o = 16; o > 0; o >>= 1) mx = fmaxf(mx, __shfl_xor(mx, o, 32));
  float e = expf(v - mx);
  float sm = e;
  for (int o = 16; o > 0; o >>= 1) sm += __shfl_xor(sm, o, 32);
  attn[row * 32 + lane] = (_Float16)(e / sm);
}

// v replicated per window (reference gc_rep tile quirk: gc[bn % B]);
// B-side of out GEMM: per batch (bn,h): Kd=k (32), N=d (64); fragment-packed.
__global__ void vrep_kernel(const float* __restrict__ gc, const float* __restrict__ vw,
                            const float* __restrict__ vb, _Float16* __restrict__ vrep) {
  int idx = blockIdx.x * 256 + threadIdx.x;
  if (idx >= 32 * 8 * 32 * 64) return;
  int d = idx & 63, k = (idx >> 6) & 31;
  int h = (idx >> 11) & 7, bn = idx >> 14;
  int o = h * 64 + d;
  const float* g = gc + ((long long)(bn % cB) * cK + k) * cC;
  const float* w = vw + (long long)o * cC;
  float acc = vb[o];
  for (int c = 0; c < cC; ++c) acc += g[c] * w[c];
  vrep[(long long)(bn * 8 + h) * 2048 + packB_idx(k, d, 64)] = (_Float16)acc;
}

// (bn,s,c) windows -> NHWC pixels
__global__ void unwindow_kernel(const _Float16* __restrict__ ctxw, _Float16* __restrict__ pix) {
  long long idx = (long long)blockIdx.x * 256 + threadIdx.x;
  if (idx >= (long long)cB * cH * cW * cC) return;
  int c = (int)(idx & 511); long long p = idx >> 9;
  int x = (int)(p & 127), y = (int)((p >> 7) & 127), b = (int)(p >> 14);
  int wy = y >> 5, iy = y & 31, wx = x >> 5, ix = x & 31;
  int bn = b * 16 + wy * 4 + wx, s = iy * 32 + ix;
  pix[idx] = ctxw[((long long)bn * 1024 + s) * 512 + c];
}

// ---------------- host launcher --------------------------------------------
extern "C" void kernel_launch(void* const* d_in, const int* in_sizes, int n_in,
                              void* d_out, int out_size, void* d_ws, size_t ws_size,
                              hipStream_t stream) {
  const float* x       = (const float*)d_in[0];
  const float* gc      = (const float*)d_in[1];
  const float* fd_w    = (const float*)d_in[2];
  const float* fd_b    = (const float*)d_in[3];
  const float* off_w   = (const float*)d_in[4];
  const float* off_b   = (const float*)d_in[5];
  const float* scale_w = (const float*)d_in[6];
  const float* scale_b = (const float*)d_in[7];
  const float* angle_w = (const float*)d_in[8];
  const float* angle_b = (const float*)d_in[9];
  const float* q_w     = (const float*)d_in[10];
  const float* q_b     = (const float*)d_in[11];
  const float* k_w     = (const float*)d_in[12];
  const float* k_b     = (const float*)d_in[13];
  const float* v_w     = (const float*)d_in[14];
  const float* v_b     = (const float*)d_in[15];
  const float* proj_w  = (const float*)d_in[16];
  const float* proj_b  = (const float*)d_in[17];
  const float* conv1_w = (const float*)d_in[18];
  const float* bn1_g   = (const float*)d_in[19];
  const float* bn1_b   = (const float*)d_in[20];
  const float* bn1_m   = (const float*)d_in[21];
  const float* bn1_v   = (const float*)d_in[22];
  const float* conv2_w = (const float*)d_in[23];
  const float* bn2_g   = (const float*)d_in[24];
  const float* bn2_b   = (const float*)d_in[25];
  const float* bn2_m   = (const float*)d_in[26];
  const float* bn2_v   = (const float*)d_in[27];
  float* out = (float*)d_out;
  char* ws = (char*)d_ws;

  size_t off = 0;
  auto alloc = [&](size_t bytes) { size_t r = off; off += (bytes + 255) & ~(size_t)255; return r; };
  size_t o_xT   = alloc(33554432ULL);   // x NHWC f16
  size_t o_fdT  = alloc(32768ULL);
  size_t o_kT   = alloc(524288ULL);
  size_t o_qT   = alloc(524288ULL);
  size_t o_pjT  = alloc(524288ULL);
  size_t o_w1T  = alloc(9437184ULL);
  size_t o_w2T  = alloc(4718592ULL);
  size_t o_bn1S = alloc(2048ULL);
  size_t o_bn1T = alloc(2048ULL);
  size_t o_bn2S = alloc(2048ULL);
  size_t o_bn2T = alloc(2048ULL);
  size_t o_log  = alloc(4194304ULL);    // logits -> probs (in place)
  size_t o_pool = alloc(65536ULL);
  size_t o_hp   = alloc(10240ULL);
  size_t o_grid = alloc(2097152ULL);
  size_t o_keyf = alloc(67108864ULL);   // keyf f32; later reused: dots f32 + attn f16
  size_t o_qpix = alloc(33554432ULL);   // q NHWC f16; later reused: ctx16
  size_t o_tx   = alloc(33554432ULL);   // tx f16; later: ctxw16, then h1_16
  size_t o_tp   = alloc(16777216ULL);   // tpT f16
  size_t o_lcT  = alloc(1048576ULL);
  size_t o_lc2  = alloc(1048576ULL);
  size_t o_qw   = alloc(33554432ULL);   // qw f16; later: ctxpix16
  size_t o_vr   = alloc(1048576ULL);
  (void)ws_size; (void)in_sizes; (void)n_in; (void)out_size;

  _Float16* xT16   = (_Float16*)(ws + o_xT);
  _Float16* fdT16  = (_Float16*)(ws + o_fdT);
  _Float16* kT16   = (_Float16*)(ws + o_kT);
  _Float16* qT16   = (_Float16*)(ws + o_qT);
  _Float16* pjT16  = (_Float16*)(ws + o_pjT);
  _Float16* w1T16  = (_Float16*)(ws + o_w1T);
  _Float16* w2T16  = (_Float16*)(ws + o_w2T);
  float* bn1S = (float*)(ws + o_bn1S); float* bn1T = (float*)(ws + o_bn1T);
  float* bn2S = (float*)(ws + o_bn2S); float* bn2T = (float*)(ws + o_bn2T);
  float* logits = (float*)(ws + o_log);
  float* pooled = (float*)(ws + o_pool);
  float* hp     = (float*)(ws + o_hp);
  float* gridb  = (float*)(ws + o_grid);
  float* keyf   = (float*)(ws + o_keyf);
  _Float16* qpix16 = (_Float16*)(ws + o_qpix);
  _Float16* tx16   = (_Float16*)(ws + o_tx);
  _Float16* tpT16  = (_Float16*)(ws + o_tp);
  _Float16* lcT16  = (_Float16*)(ws + o_lcT);
  _Float16* lc2_16 = (_Float16*)(ws + o_lc2);
  _Float16* qw16   = (_Float16*)(ws + o_qw);
  _Float16* vrep16 = (_Float16*)(ws + o_vr);
  // region reuse
  float*    dots    = (float*)(ws + o_keyf);
  _Float16* attn16  = (_Float16*)(ws + o_keyf + 33554432ULL);
  _Float16* ctx16   = (_Float16*)(ws + o_qpix);
  _Float16* ctxw16  = (_Float16*)(ws + o_tx);
  _Float16* ctxpix16= (_Float16*)(ws + o_qw);
  _Float16* h1_16   = (_Float16*)(ws + o_tx);

  // ---- precompute / conversions ----
  nchw_to_nhwc_f16<<<65536, 256, 0, stream>>>(x, xT16, cB, cC, cH, cW);
  wTpack_f16<<<(32 * 512 + 255) / 256, 256, 0, stream>>>(fd_w, fdT16, 32, 512);
  wTpack_f16<<<(512 * 512 + 255) / 256, 256, 0, stream>>>(k_w, kT16, 512, 512);
  wTpack_f16<<<(512 * 512 + 255) / 256, 256, 0, stream>>>(q_w, qT16, 512, 512);
  wTpack_f16<<<(512 * 512 + 255) / 256, 256, 0, stream>>>(proj_w, pjT16, 512, 512);
  convwTpack_f16<<<18432, 256, 0, stream>>>(conv1_w, w1T16, 512, 1024);
  convwTpack_f16<<<9216, 256, 0, stream>>>(conv2_w, w2T16, 512, 512);
  bn_params_kernel<<<2, 256, 0, stream>>>(bn1_g, bn1_b, bn1_m, bn1_v, bn1S, bn1T, 512);
  bn_params_kernel<<<2, 256, 0, stream>>>(bn2_g, bn2_b, bn2_m, bn2_v, bn2S, bn2T, 512);

  // ---- logits + probs ----
  gemm_wmma_kernel<<<dim3(256, 1, 1), 256, 0, stream>>>(
      xT16, fdT16, logits, fd_b, 32768, 32, 512, 512, 32,
      1, 0, 0, 0, 0, 0, 0, GF_BIAS, 1.f);
  probs_softmax_kernel<<<64, 256, 0, stream>>>(logits, cK, cH * cW);

  // ---- heads + grid ----
  pooled_kernel<<<64, 256, 0, stream>>>(x, pooled);
  heads_kernel<<<4, 64, 0, stream>>>(pooled, off_w, off_b, scale_w, scale_b, angle_w, angle_b, hp);
  grid_kernel<<<1024, 256, 0, stream>>>(hp, gridb);

  // ---- key features + sampling ----
  gemm_wmma_kernel<<<dim3(256, 4, 1), 256, 0, stream>>>(
      xT16, kT16, keyf, k_b, 32768, 512, 512, 512, 512,
      1, 0, 0, 0, 0, 0, 0, GF_BIAS, 1.f);
  sample_tx_kernel<<<dim3(256, 16, 16), 256, 0, stream>>>(keyf, gridb, tx16);
  sample_tp_kernel<<<dim3(128, 16, 16), 256, 0, stream>>>(logits, gridb, tpT16);

  // ---- lc = tp^T @ tx (batched 256), stored transposed (d,k) ----
  gemm_wmma_kernel<<<dim3(1, 1, 256), 256, 0, stream>>>(
      tpT16, tx16, lcT16, nullptr, 32, 64, 1024, 1024, 32,
      1, 0, 32768LL, 0, 65536LL, 0, 2048LL, GF_F16OUT | GF_TRANS, 1.f);
  lc2_kernel<<<2048, 256, 0, stream>>>(lcT16, lc2_16);

  // ---- q, windowed ----
  gemm_wmma_kernel<<<dim3(256, 4, 1), 256, 0, stream>>>(
      xT16, qT16, qpix16, q_b, 32768, 512, 512, 512, 512,
      1, 0, 0, 0, 0, 0, 0, GF_BIAS | GF_F16OUT, 1.f);
  qw_kernel<<<65536, 256, 0, stream>>>(qpix16, qw16);

  // ---- dots = scale * q @ lc^T (batched 256 over (bn,h)) ----
  gemm_wmma_kernel<<<dim3(8, 1, 256), 256, 0, stream>>>(
      qw16, lc2_16, dots, nullptr, 1024, 32, 64, 64, 32,
      1, 0, 65536LL, 0, 2048LL, 0, 32768LL, 0, 0.125f);
  attn_softmax_kernel<<<32768, 256, 0, stream>>>(dots, attn16);

  // ---- v (replicated per window, with reference tile quirk) ----
  vrep_kernel<<<2048, 256, 0, stream>>>(gc, v_w, v_b, vrep16);

  // ---- out = attn @ v -> ctx (bn, s, c) ----
  gemm_wmma_kernel<<<dim3(8, 1, 256), 256, 0, stream>>>(
      attn16, vrep16, ctx16, nullptr, 1024, 64, 32, 32, 512,
      8, 32768LL, 262144LL, 2048LL, 16384LL, 64LL, 524288LL, GF_F16OUT, 1.f);

  // ---- proj + un-window ----
  gemm_wmma_kernel<<<dim3(256, 4, 1), 256, 0, stream>>>(
      ctx16, pjT16, ctxw16, proj_b, 32768, 512, 512, 512, 512,
      1, 0, 0, 0, 0, 0, 0, GF_BIAS | GF_F16OUT, 1.f);
  unwindow_kernel<<<65536, 256, 0, stream>>>(ctxw16, ctxpix16);

  // ---- conv1 (concat[x,ctx], 1024->512) + BN + ReLU -> h1 (NHWC f16) ----
  conv3x3_wmma_kernel<<<dim3(cB * cH, 4), 256, 0, stream>>>(
      xT16, ctxpix16, w1T16, bn1S, bn1T, nullptr, h1_16, 1024, cB, cH, cW, 512);

  // ---- conv2 (512->512) + BN + ReLU -> output f32 NCHW ----
  conv3x3_wmma_kernel<<<dim3(cB * cH, 4), 256, 0, stream>>>(
      h1_16, nullptr, w2T16, bn2S, bn2T, out, nullptr, 512, cB, cH, cW, 512);
}
